// MHIMModel_35648228557601
// MI455X (gfx1250) — matrix operand order repaired and checked by reading, not compiled, not run on hardware
//
#include <hip/hip_runtime.h>

// ---------------------------------------------------------------------------
// MHIM recommender pipeline for gfx1250 (MI455X).
// Heavy GEMMs use v_wmma_f32_16x16x32_bf16 (bf16 in, f32 accumulate).
// All GEMM row counts are multiples of 16 -> no row guards in WMMA kernels.
// ---------------------------------------------------------------------------

#define Dn   128
#define Rn   12
#define NBn  8
#define Nn_  100000
#define En_  1000000
#define Pn_  400000
#define HEn  20000
#define Bn   64
#define LRn  64
#define LCn  32
#define NHn  8
#define HDn  16

typedef __bf16 bf16_t;
typedef __attribute__((ext_vector_type(16))) __bf16 bf16x16;
typedef __attribute__((ext_vector_type(8)))  __bf16 bf16x8;
typedef __attribute__((ext_vector_type(8)))  float  f32x8;

// A fragment (16-bit A 16x32): lanes<16 hold K 0..15,
// lanes>=16 K 16..31.  ka = kc*32 + (lane>=16 ? 16 : 0).
static __device__ inline bf16x16 load_a_frag(const bf16_t* base /* row start */, int ka)
{
    const bf16x8 a0 = *(const bf16x8*)(base + ka);
    const bf16x8 a1 = *(const bf16x8*)(base + ka + 8);
    return __builtin_shufflevector(a0, a1, 0, 1, 2, 3, 4, 5, 6, 7,
                                   8, 9, 10, 11, 12, 13, 14, 15);
}

// B fragment (16-bit B 32x16): lane's column, K 0..15 (lanes<16) / 16..31.
static __device__ inline bf16x16 load_b_frag(const bf16_t* p /* 16 contiguous */)
{
    const bf16x8 b0 = *(const bf16x8*)(p);
    const bf16x8 b1 = *(const bf16x8*)(p + 8);
    return __builtin_shufflevector(b0, b1, 0, 1, 2, 3, 4, 5, 6, 7,
                                   8, 9, 10, 11, 12, 13, 14, 15);
}

// ---------------------------------------------------------------------------
// Generic WMMA GEMM: C[M x 128] = A[M x 128] * B[128 x 128] (+bias)(+accum)
// B given row-major [k][n] (transB=0) or [n][k] (transB=1).
// Optional per-row scale 1/max(cnt[row*cntStride+cntOff],1) folded into A.
// Block: 256 threads = 8 waves; block covers 16 rows x 128 cols;
// wave w owns the 16x16 tile at columns [16w, 16w+16).  Requires M % 16 == 0.
// ---------------------------------------------------------------------------
__global__ __launch_bounds__(256)
void mhim_gemm128(const float* __restrict__ A, const float* __restrict__ Bw,
                  int transB, const float* __restrict__ bias,
                  const float* __restrict__ cnt, int cntStride, int cntOff,
                  float* __restrict__ C, int M, int accumulate)
{
    __shared__ bf16_t Bt[128 * 128];   // [n][k]
    __shared__ bf16_t As[16 * 128];    // [m][k]

    const int tid  = threadIdx.x;
    const int lane = tid & 31;
    const int wave = tid >> 5;
    const int m0   = blockIdx.x * 16;

    // Stage B into LDS as [n][k] bf16
    for (int i = tid; i < 128 * 128; i += 256) {
        const int n = i >> 7, k = i & 127;
        const float v = transB ? Bw[n * 128 + k] : Bw[k * 128 + n];
        Bt[n * 128 + k] = (bf16_t)v;
    }
    // Stage A strip (with optional RGCN row norm) into LDS bf16
    for (int i = tid; i < 16 * 128; i += 256) {
        const int r = i >> 7, k = i & 127;
        const int row = m0 + r;
        float s = 1.0f;
        if (cnt) s = 1.0f / fmaxf(cnt[(size_t)row * cntStride + cntOff], 1.0f);
        As[r * 128 + k] = (bf16_t)(A[(size_t)row * 128 + k] * s);
    }
    __syncthreads();

    const int nBase = wave * 16;
    const int nc    = lane & 15;
    const int hi    = lane >> 4;   // 0 or 1

    f32x8 c = {};
    if (accumulate) {
#pragma unroll
        for (int j = 0; j < 8; ++j)
            c[j] = C[(size_t)(m0 + j + hi * 8) * 128 + nBase + nc];
    }

    const bf16_t* aRow = As + (lane & 15) * 128;
    const bf16_t* bRow = Bt + (nBase + nc) * 128;
#pragma unroll
    for (int kc = 0; kc < 4; ++kc) {
        const bf16x16 af = load_a_frag(aRow, kc * 32 + hi * 16);
        const bf16x16 bf = load_b_frag(bRow + kc * 32 + hi * 16);
        c = __builtin_amdgcn_wmma_f32_16x16x32_bf16(false, af, false, bf,
                                                    (short)0, c, false, false);
    }

    const float bv = bias ? bias[nBase + nc] : 0.0f;
#pragma unroll
    for (int j = 0; j < 8; ++j)
        C[(size_t)(m0 + j + hi * 8) * 128 + nBase + nc] = c[j] + bv;
}

// ---------------------------------------------------------------------------
// Final scoring: out[64 x N] = user[64 x 128] @ kg_bf16^T + rec_bias.
// kg row-major [n][k] already matches WMMA B-fragment order -> stream from
// global. Block: 8 waves cover 64 rows x 32 cols (wave: mt = w&3, nt = w>>2).
// ---------------------------------------------------------------------------
__global__ __launch_bounds__(256)
void mhim_score(const float* __restrict__ user, const bf16_t* __restrict__ kgbf,
                const float* __restrict__ rec_bias, float* __restrict__ out, int Ntot)
{
    __shared__ bf16_t Us[64 * 128];
    const int tid  = threadIdx.x;
    const int lane = tid & 31;
    const int wave = tid >> 5;

    for (int i = tid; i < 64 * 128; i += 256) Us[i] = (bf16_t)user[i];
    __syncthreads();

    const int mt = wave & 3;
    const int nt = wave >> 2;
    const int n0 = blockIdx.x * 32 + nt * 16;
    const int nc = lane & 15;
    const int hi = lane >> 4;
    const int n  = n0 + nc;

    if (n + 32 < Ntot)  // speculative prefetch of next tile's kg rows
        __builtin_prefetch(kgbf + (size_t)(n + 32) * 128, 0, 0);

    const bf16_t* aRow = Us + (mt * 16 + (lane & 15)) * 128;
    const bf16_t* kRow = kgbf + (size_t)n * 128;

    f32x8 c = {};
#pragma unroll
    for (int kc = 0; kc < 4; ++kc) {
        const bf16x16 af = load_a_frag(aRow, kc * 32 + hi * 16);
        const bf16x16 bf = load_b_frag(kRow + kc * 32 + hi * 16);
        c = __builtin_amdgcn_wmma_f32_16x16x32_bf16(false, af, false, bf,
                                                    (short)0, c, false, false);
    }

    const float bv = rec_bias[n];
#pragma unroll
    for (int j = 0; j < 8; ++j)
        out[(size_t)(mt * 16 + j + hi * 8) * Ntot + n] = c[j] + bv;
}

// --------------------------- utility kernels -------------------------------

__global__ void mhim_zero(float* p, long long n)
{
    long long i = (long long)blockIdx.x * blockDim.x + threadIdx.x;
    if (i < n) p[i] = 0.0f;
}

__global__ void mhim_cvt_bf(const float* __restrict__ in, bf16_t* __restrict__ out, long long n)
{
    long long i = (long long)blockIdx.x * blockDim.x + threadIdx.x;
    if (i < n) out[i] = (bf16_t)in[i];
}

// w[r][i][o] = sum_b comp[r][b] * bases[b][i][o]
__global__ void mhim_build_w(const float* __restrict__ comp, const float* __restrict__ bases,
                             float* __restrict__ w, int total)
{
    int i = blockIdx.x * blockDim.x + threadIdx.x;
    if (i >= total) return;
    const int r  = i / (Dn * Dn);
    const int io = i % (Dn * Dn);
    float s = 0.0f;
#pragma unroll
    for (int b = 0; b < NBn; ++b)
        s += comp[r * NBn + b] * bases[(size_t)b * Dn * Dn + io];
    w[i] = s;
}

__global__ void mhim_count_edges(const int* __restrict__ dst, const int* __restrict__ et,
                                 float* __restrict__ cnt, int E)
{
    int e = blockIdx.x * blockDim.x + threadIdx.x;
    if (e >= E) return;
    atomicAdd(&cnt[(size_t)dst[e] * Rn + et[e]], 1.0f);
}

// A[dst] += x[src] for edges with etype == r  (32 threads x 4 floats per edge)
__global__ void mhim_rgcn_scatter(const float* __restrict__ x, const int* __restrict__ src,
                                  const int* __restrict__ dst, const int* __restrict__ et,
                                  float* __restrict__ Ar, int r, int E)
{
    long long tid = (long long)blockIdx.x * blockDim.x + threadIdx.x;
    const int e = (int)(tid >> 5);
    if (e >= E || et[e] != r) return;
    const int part = (int)(tid & 31);
    const float4 xv = *(const float4*)(x + (size_t)src[e] * Dn + part * 4);
    float* ap = Ar + (size_t)dst[e] * Dn + part * 4;
    atomicAdd(ap + 0, xv.x); atomicAdd(ap + 1, xv.y);
    atomicAdd(ap + 2, xv.z); atomicAdd(ap + 3, xv.w);
}

__global__ void mhim_hg_deg(const int* __restrict__ nodes, const int* __restrict__ edges,
                            float* __restrict__ d_deg, float* __restrict__ b_deg, int P)
{
    int p = blockIdx.x * blockDim.x + threadIdx.x;
    if (p >= P) return;
    atomicAdd(&b_deg[edges[p]], 1.0f);
    atomicAdd(&d_deg[nodes[p]], 1.0f);
}

// out[idx_out[p]] += in[idx_in[p]]   (rows of width 128)
__global__ void mhim_hg_scatter(const float* __restrict__ in, const int* __restrict__ idx_in,
                                const int* __restrict__ idx_out, float* __restrict__ out, int P)
{
    long long tid = (long long)blockIdx.x * blockDim.x + threadIdx.x;
    const int p = (int)(tid >> 5);
    if (p >= P) return;
    const int part = (int)(tid & 31);
    const float4 v = *(const float4*)(in + (size_t)idx_in[p] * Dn + part * 4);
    float* op = out + (size_t)idx_out[p] * Dn + part * 4;
    atomicAdd(op + 0, v.x); atomicAdd(op + 1, v.y);
    atomicAdd(op + 2, v.z); atomicAdd(op + 3, v.w);
}

// data[row] *= (deg[row] > 0 ? 1/deg : 0); += bias (optional)
__global__ void mhim_scale_rows(float* __restrict__ data, const float* __restrict__ deg,
                                const float* __restrict__ bias, long long total)
{
    long long i = (long long)blockIdx.x * blockDim.x + threadIdx.x;
    if (i >= total) return;
    const long long row = i >> 7;
    const int d = (int)(i & 127);
    const float dg = deg[row];
    const float s = (dg > 0.0f) ? (1.0f / dg) : 0.0f;
    data[i] = data[i] * s + (bias ? bias[d] : 0.0f);
}

__global__ void mhim_gather_related(const float* __restrict__ sess, const float* __restrict__ know,
                                    const int* __restrict__ sidx, const int* __restrict__ kidx,
                                    float* __restrict__ related)
{
    int i = blockIdx.x * blockDim.x + threadIdx.x;
    if (i >= Bn * 128 * Dn) return;
    const int b = i / (128 * Dn);
    const int rem = i % (128 * Dn);
    const int l = rem >> 7;
    const int d = rem & 127;
    int node; const float* buf;
    if (l < LRn) { node = sidx[b * LRn + l];         buf = sess; }
    else         { node = kidx[b * LRn + (l - LRn)]; buf = know; }
    related[i] = buf[(size_t)node * Dn + d];
}

__global__ void mhim_gather_context(const float* __restrict__ kg, const int* __restrict__ cidx,
                                    float* __restrict__ ctx)
{
    int i = blockIdx.x * blockDim.x + threadIdx.x;
    if (i >= Bn * LCn * Dn) return;
    const int bl = i >> 7;
    const int d = i & 127;
    ctx[i] = kg[(size_t)cidx[bl] * Dn + d];
}

// Per-(batch,head) scaled-dot-product attention: 32 queries x 128 keys.
__global__ __launch_bounds__(128)
void mhim_mha_attn(const float* __restrict__ qh, const float* __restrict__ kh,
                   const float* __restrict__ vh, float* __restrict__ o)
{
    const int b = blockIdx.x >> 3;
    const int h = blockIdx.x & 7;
    __shared__ float e[LCn][128];
    const int t = threadIdx.x;          // key index
    const float scale = 0.25f;          // 1/sqrt(HD=16)

    float kv[HDn];
    const float* kp = kh + ((size_t)(b * 128 + t)) * Dn + h * HDn;
#pragma unroll
    for (int d = 0; d < HDn; ++d) kv[d] = kp[d];

    for (int q = 0; q < LCn; ++q) {
        const float* qp = qh + ((size_t)(b * LCn + q)) * Dn + h * HDn;
        float s = 0.0f;
#pragma unroll
        for (int d = 0; d < HDn; ++d) s += qp[d] * kv[d];
        e[q][t] = s * scale;
    }
    __syncthreads();

    if (t < LCn) {                      // softmax row t
        float m = -1e30f;
        for (int k = 0; k < 128; ++k) m = fmaxf(m, e[t][k]);
        float s = 0.0f;
        for (int k = 0; k < 128; ++k) { float w = __expf(e[t][k] - m); e[t][k] = w; s += w; }
        const float inv = 1.0f / s;
        for (int k = 0; k < 128; ++k) e[t][k] *= inv;
    }
    __syncthreads();

    const int q = t >> 2;
    const int d0 = (t & 3) * 4;
    float acc[4] = {0, 0, 0, 0};
    for (int k = 0; k < 128; ++k) {
        const float w = e[q][k];
        const float* vp = vh + ((size_t)(b * 128 + k)) * Dn + h * HDn + d0;
#pragma unroll
        for (int i = 0; i < 4; ++i) acc[i] += w * vp[i];
    }
    float* op = o + ((size_t)(b * LCn + q)) * Dn + h * HDn + d0;
#pragma unroll
    for (int i = 0; i < 4; ++i) op[i] = acc[i];
}

// e[row] = sum_d tanh(hA[row][d]) * bvec[d]
__global__ __launch_bounds__(128)
void mhim_tanh_dot(const float* __restrict__ hA, const float* __restrict__ bvec,
                   float* __restrict__ e, int rows)
{
    const int row = blockIdx.x;
    if (row >= rows) return;
    const int t = threadIdx.x;
    __shared__ float red[128];
    red[t] = tanhf(hA[(size_t)row * Dn + t]) * bvec[t];
    __syncthreads();
    for (int s = 64; s > 0; s >>= 1) {
        if (t < s) red[t] += red[t + s];
        __syncthreads();
    }
    if (t == 0) e[row] = red[0];
}

// out[b][d] = sum_l softmax_l(e[b])[l] * h[b][l][d]
__global__ __launch_bounds__(128)
void mhim_attn_pool(const float* __restrict__ e, const float* __restrict__ h,
                    float* __restrict__ out, int L)
{
    const int b = blockIdx.x;
    const int t = threadIdx.x;          // feature dim
    float m = -1e30f;
    for (int l = 0; l < L; ++l) m = fmaxf(m, e[b * L + l]);
    float s = 0.0f;
    for (int l = 0; l < L; ++l) s += __expf(e[b * L + l] - m);
    const float inv = 1.0f / s;
    float acc = 0.0f;
    for (int l = 0; l < L; ++l)
        acc += __expf(e[b * L + l] - m) * inv * h[((size_t)(b * L + l)) * Dn + t];
    out[b * Dn + t] = acc;
}

__global__ void mhim_build_usercat(const float* __restrict__ ctx, const float* __restrict__ his,
                                   float* __restrict__ uc)
{
    int i = blockIdx.x * blockDim.x + threadIdx.x;
    if (i >= Bn * 33 * Dn) return;
    const int row = i >> 7;
    const int d = i & 127;
    const int b = row / 33;
    const int l = row % 33;
    uc[i] = (l < LCn) ? ctx[((size_t)(b * LCn + l)) * Dn + d] : his[b * Dn + d];
}

// ---------------------------------------------------------------------------
extern "C" void kernel_launch(void* const* d_in, const int* in_sizes, int n_in,
                              void* d_out, int out_size, void* d_ws, size_t ws_size,
                              hipStream_t stream)
{
    const float* entity_emb = (const float*)d_in[0];
    const float* rgcn_bases = (const float*)d_in[1];
    const float* rgcn_comp  = (const float*)d_in[2];
    const float* rgcn_root  = (const float*)d_in[3];
    const float* rgcn_bias  = (const float*)d_in[4];
    const float* sess_theta = (const float*)d_in[5];
    const float* sess_bias  = (const float*)d_in[6];
    const float* know_theta = (const float*)d_in[7];
    const float* know_bias  = (const float*)d_in[8];
    const float* in_proj_w  = (const float*)d_in[9];
    const float* in_proj_b  = (const float*)d_in[10];
    const float* out_proj_w = (const float*)d_in[11];
    const float* out_proj_b = (const float*)d_in[12];
    const float* attn_his_a = (const float*)d_in[13];
    const float* attn_his_b = (const float*)d_in[14];
    const float* attn_a     = (const float*)d_in[15];
    const float* attn_b     = (const float*)d_in[16];
    const float* rec_bias   = (const float*)d_in[17];
    const int* edge_src     = (const int*)d_in[18];
    const int* edge_dst     = (const int*)d_in[19];
    const int* edge_type    = (const int*)d_in[20];
    const int* sess_nodes   = (const int*)d_in[21];
    const int* sess_edges   = (const int*)d_in[22];
    const int* know_nodes   = (const int*)d_in[23];
    const int* know_edges   = (const int*)d_in[24];
    const int* sess_rel_idx = (const int*)d_in[25];
    const int* know_rel_idx = (const int*)d_in[26];
    const int* context_idx  = (const int*)d_in[27];
    (void)in_sizes; (void)n_in; (void)out_size; (void)ws_size;

    // --- workspace carve ---
    char* wp = (char*)d_ws;
    auto alloc = [&](size_t bytes) -> void* {
        void* q = (void*)wp; wp += (bytes + 255) & ~(size_t)255; return q;
    };
    float*  w_all   = (float*)alloc((size_t)Rn * Dn * Dn * 4);
    float*  cnt     = (float*)alloc((size_t)Nn_ * Rn * 4);
    float*  kg      = (float*)alloc((size_t)Nn_ * Dn * 4);
    float*  Ar      = (float*)alloc((size_t)Nn_ * Dn * 4);
    bf16_t* kgbf    = (bf16_t*)alloc((size_t)Nn_ * Dn * 2);
    float*  xt      = (float*)alloc((size_t)Nn_ * Dn * 4);
    float*  ef      = (float*)alloc((size_t)HEn * Dn * 4);
    float*  b_deg   = (float*)alloc((size_t)HEn * 4);
    float*  d_deg   = (float*)alloc((size_t)Nn_ * 4);
    float*  sessb   = (float*)alloc((size_t)Nn_ * Dn * 4);
    float*  knowb   = (float*)alloc((size_t)Nn_ * Dn * 4);
    float*  related = (float*)alloc((size_t)Bn * 128 * Dn * 4);
    float*  ctx     = (float*)alloc((size_t)Bn * LCn * Dn * 4);
    float*  qh      = (float*)alloc((size_t)Bn * LCn * Dn * 4);
    float*  kh      = (float*)alloc((size_t)Bn * 128 * Dn * 4);
    float*  vh      = (float*)alloc((size_t)Bn * 128 * Dn * 4);
    float*  obuf    = (float*)alloc((size_t)Bn * LCn * Dn * 4);
    float*  att_rel = (float*)alloc((size_t)Bn * LCn * Dn * 4);
    float*  hA      = (float*)alloc((size_t)Bn * 33 * Dn * 4);
    float*  ebuf    = (float*)alloc((size_t)Bn * 33 * 4);
    float*  his     = (float*)alloc((size_t)Bn * Dn * 4);
    float*  ucat    = (float*)alloc((size_t)Bn * 33 * Dn * 4);
    float*  user    = (float*)alloc((size_t)Bn * Dn * 4);
    float*  out     = (float*)d_out;

    auto blocks = [](long long n, int bs) { return (unsigned)((n + bs - 1) / bs); };

    // --- relation weights + edge-pair counts ---
    mhim_build_w<<<blocks(Rn * Dn * Dn, 256), 256, 0, stream>>>(rgcn_comp, rgcn_bases, w_all, Rn * Dn * Dn);
    mhim_zero<<<blocks((long long)Nn_ * Rn, 256), 256, 0, stream>>>(cnt, (long long)Nn_ * Rn);
    mhim_count_edges<<<blocks(En_, 256), 256, 0, stream>>>(edge_dst, edge_type, cnt, En_);

    // --- RGCN: kg = x @ root + bias ---
    mhim_gemm128<<<blocks(Nn_, 16), 256, 0, stream>>>(entity_emb, rgcn_root, 0, rgcn_bias,
                                                      nullptr, 0, 0, kg, Nn_, 0);
    // --- per-relation: scatter + normalized GEMM accumulate ---
    for (int r = 0; r < Rn; ++r) {
        mhim_zero<<<blocks((long long)Nn_ * Dn, 256), 256, 0, stream>>>(Ar, (long long)Nn_ * Dn);
        mhim_rgcn_scatter<<<blocks((long long)En_ * 32, 256), 256, 0, stream>>>(
            entity_emb, edge_src, edge_dst, edge_type, Ar, r, En_);
        mhim_gemm128<<<blocks(Nn_, 16), 256, 0, stream>>>(Ar, w_all + (size_t)r * Dn * Dn, 0, nullptr,
                                                          cnt, Rn, r, kg, Nn_, 1);
    }
    mhim_cvt_bf<<<blocks((long long)Nn_ * Dn, 256), 256, 0, stream>>>(kg, kgbf, (long long)Nn_ * Dn);

    // --- hypergraph convolutions (sess, know) ---
    const int* hn[2] = { sess_nodes, know_nodes };
    const int* he[2] = { sess_edges, know_edges };
    const float* hth[2] = { sess_theta, know_theta };
    const float* hb[2]  = { sess_bias,  know_bias };
    float* hout[2] = { sessb, knowb };
    for (int g = 0; g < 2; ++g) {
        mhim_gemm128<<<blocks(Nn_, 16), 256, 0, stream>>>(kg, hth[g], 0, nullptr,
                                                          nullptr, 0, 0, xt, Nn_, 0);
        mhim_zero<<<blocks(HEn, 256), 256, 0, stream>>>(b_deg, HEn);
        mhim_zero<<<blocks(Nn_, 256), 256, 0, stream>>>(d_deg, Nn_);
        mhim_zero<<<blocks((long long)HEn * Dn, 256), 256, 0, stream>>>(ef, (long long)HEn * Dn);
        mhim_zero<<<blocks((long long)Nn_ * Dn, 256), 256, 0, stream>>>(hout[g], (long long)Nn_ * Dn);
        mhim_hg_deg<<<blocks(Pn_, 256), 256, 0, stream>>>(hn[g], he[g], d_deg, b_deg, Pn_);
        mhim_hg_scatter<<<blocks((long long)Pn_ * 32, 256), 256, 0, stream>>>(xt, hn[g], he[g], ef, Pn_);
        mhim_scale_rows<<<blocks((long long)HEn * Dn, 256), 256, 0, stream>>>(ef, b_deg, nullptr,
                                                                              (long long)HEn * Dn);
        mhim_hg_scatter<<<blocks((long long)Pn_ * 32, 256), 256, 0, stream>>>(ef, he[g], hn[g], hout[g], Pn_);
        mhim_scale_rows<<<blocks((long long)Nn_ * Dn, 256), 256, 0, stream>>>(hout[g], d_deg, hb[g],
                                                                              (long long)Nn_ * Dn);
    }

    // --- gathers ---
    mhim_gather_related<<<blocks((long long)Bn * 128 * Dn, 256), 256, 0, stream>>>(
        sessb, knowb, sess_rel_idx, know_rel_idx, related);
    mhim_gather_context<<<blocks((long long)Bn * LCn * Dn, 256), 256, 0, stream>>>(kg, context_idx, ctx);

    // --- MHA: projections (WMMA) + small attention ---
    mhim_gemm128<<<blocks(Bn * LCn, 16), 256, 0, stream>>>(ctx, in_proj_w, 1, in_proj_b,
                                                           nullptr, 0, 0, qh, Bn * LCn, 0);
    mhim_gemm128<<<blocks(Bn * 128, 16), 256, 0, stream>>>(related, in_proj_w + Dn * Dn, 1, in_proj_b + Dn,
                                                           nullptr, 0, 0, kh, Bn * 128, 0);
    mhim_gemm128<<<blocks(Bn * 128, 16), 256, 0, stream>>>(related, in_proj_w + 2 * Dn * Dn, 1, in_proj_b + 2 * Dn,
                                                           nullptr, 0, 0, vh, Bn * 128, 0);
    mhim_mha_attn<<<Bn * NHn, 128, 0, stream>>>(qh, kh, vh, obuf);
    mhim_gemm128<<<blocks(Bn * LCn, 16), 256, 0, stream>>>(obuf, out_proj_w, 1, out_proj_b,
                                                           nullptr, 0, 0, att_rel, Bn * LCn, 0);

    // --- self attention over history (L = 32) ---
    mhim_gemm128<<<blocks(Bn * LCn, 16), 256, 0, stream>>>(att_rel, attn_his_a, 0, nullptr,
                                                           nullptr, 0, 0, hA, Bn * LCn, 0);
    mhim_tanh_dot<<<Bn * LCn, 128, 0, stream>>>(hA, attn_his_b, ebuf, Bn * LCn);
    mhim_attn_pool<<<Bn, 128, 0, stream>>>(ebuf, att_rel, his, LCn);

    // --- user attention over [context; his] (L = 33) ---
    mhim_build_usercat<<<blocks((long long)Bn * 33 * Dn, 256), 256, 0, stream>>>(ctx, his, ucat);
    mhim_gemm128<<<blocks(Bn * 33, 16), 256, 0, stream>>>(ucat, attn_a, 0, nullptr,
                                                          nullptr, 0, 0, hA, Bn * 33, 0);
    mhim_tanh_dot<<<Bn * 33, 128, 0, stream>>>(hA, attn_b, ebuf, Bn * 33);
    mhim_attn_pool<<<Bn, 128, 0, stream>>>(ebuf, ucat, user, 33);

    // --- final scoring: out = user @ kg^T + rec_bias ---
    mhim_score<<<blocks(Nn_, 32), 256, 0, stream>>>(user, kgbf, rec_bias, out, Nn_);
}
